// CombinedGraphLayer_67791763800346
// MI455X (gfx1250) — matrix operand order repaired and checked by reading, not compiled, not run on hardware
//
#include <hip/hip_runtime.h>

// ---------------------------------------------------------------------------
// Problem constants (from reference setup_inputs)
// ---------------------------------------------------------------------------
#define BB     8
#define NN     12800
#define FF     256
#define DD     128
#define BINSZ  128
#define NBINS  100          // N / BIN_SIZE
#define NHALF  50           // n_bins / 2
#define ROTC   100          // MAX_BINS/2 (rotations column stride)
#define BNROWS (BB * NN)    // 102400

typedef __attribute__((ext_vector_type(16))) __bf16 v16bf;
typedef __attribute__((ext_vector_type(8)))  __bf16 v8bf;
typedef __attribute__((ext_vector_type(8)))  float  v8f;
typedef __attribute__((ext_vector_type(4))) unsigned int u32x4;
typedef __attribute__((ext_vector_type(8))) int          i32x8;
typedef __attribute__((ext_vector_type(4))) int          i32x4;

// ---------------------------------------------------------------------------
// WMMA helper: D = A(16x32 bf16) * B(32x16 bf16) + C(16x16 f32)
// ---------------------------------------------------------------------------
__device__ __forceinline__ v8f wmma_bf16(v16bf a, v16bf b, v8f c) {
  return __builtin_amdgcn_wmma_f32_16x16x32_bf16(false, a, false, b,
                                                 (short)0, c, false, false);
}

// A-fragment (16x32 bf16): lane holds row M=lane&15.
// elems e -> K = k0 + 16*lhi + e (contiguous)
__device__ __forceinline__ v16bf load_a_frag(const __bf16* rowk0, int lhi) {
  v8bf lo = *(const v8bf*)(rowk0 + lhi * 16);
  v8bf hi = *(const v8bf*)(rowk0 + lhi * 16 + 8);
  v16bf r;
#pragma unroll
  for (int i = 0; i < 8; i++) { r[i] = lo[i]; r[8 + i] = hi[i]; }
  return r;
}

// B-fragment (32x16 bf16), B stored K-contiguous per column:
// lane holds col N=lane&15; elems e -> K = k0 + 16*lhi + e (contiguous).
__device__ __forceinline__ v16bf load_b_frag(const __bf16* colk0, int lhi) {
  v8bf lo = *(const v8bf*)(colk0 + lhi * 16);
  v8bf hi = *(const v8bf*)(colk0 + lhi * 16 + 8);
  v16bf r;
#pragma unroll
  for (int i = 0; i < 8; i++) { r[i] = lo[i]; r[8 + i] = hi[i]; }
  return r;
}

// Build a fragment from two fp32 row segments (converted to bf16)
__device__ __forceinline__ v16bf cvt_frag(const float* p0, const float* p1) {
  v16bf r;
#pragma unroll
  for (int i = 0; i < 8; i++) { r[i] = (__bf16)p0[i]; r[8 + i] = (__bf16)p1[i]; }
  return r;
}

__device__ __forceinline__ float elu_f(float v) {
  return v > 0.f ? v : (__expf(v) - 1.f);
}

// ---------------------------------------------------------------------------
// K0: fp32 [K][Nc]  ->  bf16 transposed [Nc][K]  (weights, K-contiguous cols)
// ---------------------------------------------------------------------------
__global__ void convT_kernel(const float* __restrict__ W,
                             __bf16* __restrict__ Wt, int K, int Nc) {
  int idx = blockIdx.x * 256 + threadIdx.x;
  if (idx >= K * Nc) return;
  int k = idx / Nc, c = idx % Nc;
  Wt[(size_t)c * K + k] = (__bf16)W[idx];
}

// ---------------------------------------------------------------------------
// K1: LayerNorm over F=256, one wave per row, output bf16
// ---------------------------------------------------------------------------
__global__ __launch_bounds__(256)
void ln_kernel(const float* __restrict__ x, const float* __restrict__ gamma,
               const float* __restrict__ beta, __bf16* __restrict__ xnb) {
  int row  = blockIdx.x * 8 + (threadIdx.x >> 5);
  int lane = threadIdx.x & 31;
  if (row >= BNROWS) return;
  const float* xr = x + (size_t)row * FF;
  float vals[8], s = 0.f, s2 = 0.f;
#pragma unroll
  for (int i = 0; i < 8; i++) {
    float v = xr[lane + i * 32];
    vals[i] = v; s += v; s2 += v * v;
  }
#pragma unroll
  for (int m = 16; m >= 1; m >>= 1) {
    s  += __shfl_xor(s,  m, 32);
    s2 += __shfl_xor(s2, m, 32);
  }
  float mu  = s * (1.f / FF);
  float var = s2 * (1.f / FF) - mu * mu;
  float rs  = rsqrtf(var + 1e-6f);
#pragma unroll
  for (int i = 0; i < 8; i++) {
    int c = lane + i * 32;
    xnb[(size_t)row * FF + c] = (__bf16)((vals[i] - mu) * rs * gamma[c] + beta[c]);
  }
}

// ---------------------------------------------------------------------------
// K2: GEMM  C[M,Nc] = act(A[M,256] @ B[256,Nc] + bias)
//     A bf16 row-major, Bt bf16 transposed [Nc][256].
//     The 16x256 bf16 A slab (8 KB) shared by all 8 waves of the block is
//     staged into LDS once via the Tensor Data Mover (TDM) and read back as
//     16-byte ds loads; B-fragments stream from L2-resident weights.
//     act: 0 none, 1 elu, 2 sigmoid. Grid is sized so col0 < Nc always.
// ---------------------------------------------------------------------------
__global__ __launch_bounds__(256)
void gemm_k256(const __bf16* __restrict__ A, const __bf16* __restrict__ Bt,
               const float* __restrict__ bias, __bf16* __restrict__ outb,
               float* __restrict__ outf, int Nc, int act) {
  __shared__ __attribute__((aligned(16))) __bf16 sA[16 * 256];  // 8 KB
  int wave = threadIdx.x >> 5, lane = threadIdx.x & 31;
  int lrow = lane & 15, lhi = lane >> 4;
  int row0 = blockIdx.x * 16;
  int col0 = (blockIdx.y * 8 + wave) * 16;

#if __has_builtin(__builtin_amdgcn_tensor_load_to_lds)
  if (wave == 0) {  // wave-uniform; tensor ops ignore EXEC, issue once per wave
    unsigned lds_off = (unsigned)(size_t)(void*)sA;  // LDS aperture: low 32 bits
    unsigned long long ga =
        (unsigned long long)(size_t)(const void*)(A + (size_t)row0 * 256);
    u32x4 g0;
    g0[0] = 1u;                                  // count=1, user descriptor
    g0[1] = lds_off;                             // lds_addr
    g0[2] = (unsigned)(ga & 0xFFFFFFFFull);      // global_addr[31:0]
    g0[3] = ((unsigned)(ga >> 32) & 0x1FFFFFFu)  // global_addr[56:32]
            | 0x80000000u;                       // type=2 ("image")
    i32x8 g1;
    g1[0] = (int)(1u << 16);        // data_size = 1 -> 2 bytes (bf16)
    g1[1] = (int)(256u << 16);      // tensor_dim0 = 256 (lo16)
    g1[2] = (int)(16u << 16);       // tensor_dim0 hi=0 | tensor_dim1 = 16
    g1[3] = (int)(256u << 16);      // tensor_dim1 hi=0 | tile_dim0 = 256
    g1[4] = 16;                     // tile_dim1 = 16, tile_dim2 = 0
    g1[5] = 256;                    // tensor_dim0_stride = 256 (lo32)
    g1[6] = 0;                      // stride0 hi | stride1 lo
    g1[7] = 0;                      // stride1 hi
    i32x4 gz = {0, 0, 0, 0};
#if defined(__clang_major__) && __clang_major__ >= 23
    i32x8 gz8 = {0, 0, 0, 0, 0, 0, 0, 0};
    __builtin_amdgcn_tensor_load_to_lds(g0, g1, gz, gz, gz8, 0);
#else
    __builtin_amdgcn_tensor_load_to_lds(g0, g1, gz, gz, 0);
#endif
    __builtin_amdgcn_s_wait_tensorcnt(0);
  }
#else
  {  // fallback: cooperative staged copy, 32 bytes per thread
    const __bf16* src = A + (size_t)row0 * 256;
    v8bf* d = (v8bf*)sA;
    const v8bf* s = (const v8bf*)src;
    d[threadIdx.x * 2 + 0] = s[threadIdx.x * 2 + 0];
    d[threadIdx.x * 2 + 1] = s[threadIdx.x * 2 + 1];
  }
#endif
  __syncthreads();

  const __bf16* Arow = sA + (size_t)lrow * 256;
  const __bf16* Bcol = Bt + (size_t)(col0 + lrow) * 256;
  v8f acc = {};
#pragma unroll
  for (int ks = 0; ks < 8; ks++) {
    v16bf a = load_a_frag(Arow + ks * 32, lhi);
    v16bf b = load_b_frag(Bcol + ks * 32, lhi);
    acc = wmma_bf16(a, b, acc);
  }
  int col = col0 + lrow;
  float bv = bias ? bias[col] : 0.f;
#pragma unroll
  for (int r = 0; r < 8; r++) {
    int row = row0 + r + lhi * 8;
    float v = acc[r] + bv;
    if (act == 1) v = elu_f(v);
    else if (act == 2) v = 1.f / (1.f + __expf(-v));
    size_t off = (size_t)row * Nc + col;
    if (outb) outb[off] = (__bf16)v;
    if (outf) outf[off] = v;
  }
}

// ---------------------------------------------------------------------------
// K3: LSH projection + argmax over [mul, -mul] (first-max tie-break)
// ---------------------------------------------------------------------------
__global__ __launch_bounds__(256)
void lsh_kernel(const float* __restrict__ xdist, const float* __restrict__ rot,
                const unsigned char* __restrict__ msk, int* __restrict__ bin_idx) {
  int i = blockIdx.x * 256 + threadIdx.x;
  if (i >= BNROWS) return;
  float acc[NHALF];
#pragma unroll
  for (int c = 0; c < NHALF; c++) acc[c] = 0.f;
  const float* xr = xdist + (size_t)i * DD;
  for (int d = 0; d < DD; d++) {
    float xv = xr[d];
    const float* rr = rot + d * ROTC;
#pragma unroll
    for (int c = 0; c < NHALF; c++) acc[c] = fmaf(xv, rr[c], acc[c]);
  }
  float best = acc[0]; int bi = 0;
#pragma unroll
  for (int c = 1; c < NHALF; c++) if (acc[c] > best) { best = acc[c]; bi = c; }
#pragma unroll
  for (int c = 0; c < NHALF; c++) {
    float v = -acc[c];
    if (v > best) { best = v; bi = NHALF + c; }
  }
  if (!msk[i]) bi += (NBINS - 1);
  bin_idx[i] = bi;
}

// ---------------------------------------------------------------------------
// K4: stable counting sort per batch (deterministic == jnp.argsort order)
// ---------------------------------------------------------------------------
#define SORT_T  64
#define SORT_NB 200
__global__ __launch_bounds__(SORT_T)
void sort_kernel(const int* __restrict__ bin_idx, int* __restrict__ flat) {
  __shared__ int hist[SORT_T * SORT_NB];
  __shared__ int tot[SORT_NB];
  __shared__ int base[SORT_NB];
  int b = blockIdx.x, t = threadIdx.x;
  for (int i = t; i < SORT_T * SORT_NB; i += SORT_T) hist[i] = 0;
  __syncthreads();
  const int CH = NN / SORT_T;  // 200
  const int* bi = bin_idx + (size_t)b * NN;
  int start = t * CH;
  for (int i = 0; i < CH; i++) hist[t * SORT_NB + bi[start + i]]++;
  __syncthreads();
  for (int bn = t; bn < SORT_NB; bn += SORT_T) {
    int s = 0;
    for (int c = 0; c < SORT_T; c++) s += hist[c * SORT_NB + bn];
    tot[bn] = s;
  }
  __syncthreads();
  if (t == 0) {
    int run = 0;
    for (int bn = 0; bn < SORT_NB; bn++) { base[bn] = run; run += tot[bn]; }
  }
  __syncthreads();
  for (int bn = t; bn < SORT_NB; bn += SORT_T) {
    int run = base[bn];
    for (int c = 0; c < SORT_T; c++) {
      int tmp = hist[c * SORT_NB + bn];
      hist[c * SORT_NB + bn] = run;
      run += tmp;
    }
  }
  __syncthreads();
  int* fo = flat + (size_t)b * NN;
  for (int i = 0; i < CH; i++) {
    int idx = start + i;
    int pos = hist[t * SORT_NB + bi[idx]]++;
    fo[pos] = idx;  // flat[p] = original index (stable)
  }
}

// ---------------------------------------------------------------------------
// K5: per-(batch,bin) kernel — Gram(WMMA) -> dm -> degrees -> dm@fhp(WMMA)
//     -> gated-elu epilogue -> scatter to d_out via flat indices.
// ---------------------------------------------------------------------------
#define DMST 136  // padded bf16 row stride for dm in LDS (mult of 8)
__global__ __launch_bounds__(256)
void bin_kernel(const float* __restrict__ xdist, const __bf16* __restrict__ fhp,
                const __bf16* __restrict__ gateb, const __bf16* __restrict__ fhetb,
                const int* __restrict__ flat, const unsigned char* __restrict__ msk,
                float* __restrict__ out) {
  __shared__ __attribute__((aligned(16))) __bf16 sdm[BINSZ * DMST];
  __shared__ float sna[BINSZ];
  __shared__ float snorm[BINSZ];
  __shared__ float smsk[BINSZ];
  __shared__ int   sidx[BINSZ];

  int b = blockIdx.x / NBINS;
  int bin = blockIdx.x % NBINS;
  int t = threadIdx.x;
  int wave = t >> 5, lane = t & 31;
  int lrow = lane & 15, lhi = lane >> 4;

  // Stage 1: gather indices, mask, squared row norms (fp32)
  if (t < BINSZ) {
    int g = flat[(size_t)b * NN + bin * BINSZ + t];
    sidx[t] = g;
    smsk[t] = msk[(size_t)b * NN + g] ? 1.f : 0.f;
    const float* xr = xdist + (size_t)(b * NN + g) * DD;
    float s = 0.f;
    for (int d = 0; d < DD; d++) s += xr[d] * xr[d];
    sna[t] = s;
  }
  __syncthreads();

  // Stage 2: Gram matrix via WMMA -> Gaussian kernel dm into LDS (bf16)
  for (int tile = wave; tile < 64; tile += 8) {
    int tm = tile >> 3, tn = tile & 7;
    const float* Ap = xdist + (size_t)(b * NN + sidx[tm * 16 + lrow]) * DD;
    const float* Bp = xdist + (size_t)(b * NN + sidx[tn * 16 + lrow]) * DD;
    v8f acc = {};
#pragma unroll
    for (int ks = 0; ks < 4; ks++) {
      int k0 = ks * 32;
      v16bf a = cvt_frag(Ap + k0 + lhi * 16, Ap + k0 + lhi * 16 + 8);
      v16bf bf = cvt_frag(Bp + k0 + lhi * 16, Bp + k0 + lhi * 16 + 8);
      acc = wmma_bf16(a, bf, acc);
    }
#pragma unroll
    for (int r = 0; r < 8; r++) {
      int i = tm * 16 + r + lhi * 8;
      int j = tn * 16 + lrow;
      float d2 = fmaxf(sna[i] + sna[j] - 2.f * acc[r], 1e-6f);
      float dmv = __expf(-0.1f * sqrtf(d2));
      dmv = fminf(fmaxf(dmv, 0.f), 1.f) * smsk[i] * smsk[j];
      sdm[i * DMST + j] = (__bf16)dmv;
    }
  }
  __syncthreads();

  // Stage 3: degrees -> norm
  if (t < BINSZ) {
    float s = 0.f;
    for (int j = 0; j < BINSZ; j++) s += fabsf((float)sdm[t * DMST + j]);
    s = fminf(s, 1000.f);
    snorm[t] = rsqrtf(s + 1e-6f) * smsk[t];
  }
  __syncthreads();

  // Stage 4: f_hom = (dm @ (fhp * norm)) * norm, fuse gate/elu, scatter out
  for (int tile = wave; tile < 128; tile += 8) {
    int tm = tile >> 4, tn = tile & 15;
    const __bf16* drow = sdm + (tm * 16 + lrow) * DMST;
    int col = tn * 16 + lrow;
    v8f acc = {};
#pragma unroll
    for (int ks = 0; ks < 4; ks++) {
      int k0 = ks * 32;
      v16bf a = load_a_frag(drow + k0, lhi);
      v16bf bf;
#pragma unroll
      for (int e = 0; e < 16; e++) {
        int k = k0 + lhi * 16 + e;
        float v = (float)fhp[(size_t)(b * NN + sidx[k]) * FF + col];
        bf[e] = (__bf16)(v * snorm[k] * smsk[k]);
      }
      acc = wmma_bf16(a, bf, acc);
    }
#pragma unroll
    for (int r = 0; r < 8; r++) {
      int i = tm * 16 + r + lhi * 8;
      size_t off = (size_t)(b * NN + sidx[i]) * FF + col;
      float fh = acc[r] * snorm[i];
      float gt = (float)gateb[off];
      float fe = (float)fhetb[off] * smsk[i];
      out[off] = elu_f(gt * fh + (1.f - gt) * fe) * smsk[i];
    }
  }
}

// ---------------------------------------------------------------------------
// Launch
// ---------------------------------------------------------------------------
extern "C" void kernel_launch(void* const* d_in, const int* in_sizes, int n_in,
                              void* d_out, int out_size, void* d_ws, size_t ws_size,
                              hipStream_t stream) {
  (void)in_sizes; (void)n_in; (void)out_size; (void)ws_size;
  const float* x     = (const float*)d_in[0];
  const unsigned char* msk = (const unsigned char*)d_in[1];
  const float* rot   = (const float*)d_in[2];
  const float* gamma = (const float*)d_in[3];
  const float* beta  = (const float*)d_in[4];
  const float* w0 = (const float*)d_in[5];
  const float* b0 = (const float*)d_in[6];
  const float* w1 = (const float*)d_in[7];
  const float* b1 = (const float*)d_in[8];
  const float* w2 = (const float*)d_in[9];
  const float* b2 = (const float*)d_in[10];
  const float* Wt = (const float*)d_in[11];
  const float* bt = (const float*)d_in[12];
  const float* Wh = (const float*)d_in[13];
  const float* th = (const float*)d_in[14];

  // workspace carve-up (~369 MB total)
  char* ws = (char*)d_ws;
  auto alloc = [&](size_t bytes) -> char* {
    char* p = ws; ws += (bytes + 255) & ~(size_t)255; return p;
  };
  __bf16* xnb   = (__bf16*)alloc((size_t)BNROWS * FF * 2);
  __bf16* h0b   = (__bf16*)alloc((size_t)BNROWS * FF * 2);
  __bf16* h1b   = (__bf16*)alloc((size_t)BNROWS * FF * 2);
  float*  xdist = (float*) alloc((size_t)BNROWS * DD * 4);
  __bf16* gateb = (__bf16*)alloc((size_t)BNROWS * FF * 2);
  __bf16* fhetb = (__bf16*)alloc((size_t)BNROWS * FF * 2);
  __bf16* fhpb  = (__bf16*)alloc((size_t)BNROWS * FF * 2);
  int* bin_idx  = (int*)   alloc((size_t)BNROWS * 4);
  int* flat     = (int*)   alloc((size_t)BNROWS * 4);
  __bf16* w0t = (__bf16*)alloc(256 * 256 * 2);
  __bf16* w1t = (__bf16*)alloc(256 * 256 * 2);
  __bf16* w2t = (__bf16*)alloc(256 * 128 * 2);
  __bf16* Wtt = (__bf16*)alloc(256 * 256 * 2);
  __bf16* Wht = (__bf16*)alloc(256 * 256 * 2);
  __bf16* tht = (__bf16*)alloc(256 * 256 * 2);

  // weights -> bf16 transposed
  convT_kernel<<<(256 * 256 + 255) / 256, 256, 0, stream>>>(w0, w0t, 256, 256);
  convT_kernel<<<(256 * 256 + 255) / 256, 256, 0, stream>>>(w1, w1t, 256, 256);
  convT_kernel<<<(256 * 128 + 255) / 256, 256, 0, stream>>>(w2, w2t, 256, 128);
  convT_kernel<<<(256 * 256 + 255) / 256, 256, 0, stream>>>(Wt, Wtt, 256, 256);
  convT_kernel<<<(256 * 256 + 255) / 256, 256, 0, stream>>>(Wh, Wht, 256, 256);
  convT_kernel<<<(256 * 256 + 255) / 256, 256, 0, stream>>>(th, tht, 256, 256);

  // LayerNorm
  ln_kernel<<<BNROWS / 8, 256, 0, stream>>>(x, gamma, beta, xnb);

  // FFN + GHConv linear maps (WMMA GEMMs, A tile staged via TDM)
  dim3 g2(BNROWS / 16, 2), g1(BNROWS / 16, 1);
  gemm_k256<<<g2, 256, 0, stream>>>(xnb, w0t, b0, h0b, nullptr, 256, 1);       // elu
  gemm_k256<<<g2, 256, 0, stream>>>(h0b, w1t, b1, h1b, nullptr, 256, 1);       // elu
  gemm_k256<<<g1, 256, 0, stream>>>(h1b, w2t, b2, nullptr, xdist, 128, 0);     // x_dist f32
  gemm_k256<<<g2, 256, 0, stream>>>(xnb, Wtt, bt, gateb, nullptr, 256, 2);     // sigmoid gate
  gemm_k256<<<g2, 256, 0, stream>>>(xnb, Wht, nullptr, fhetb, nullptr, 256, 0);// f_het
  gemm_k256<<<g2, 256, 0, stream>>>(xnb, tht, nullptr, fhpb, nullptr, 256, 0); // f_hom_pre

  // LSH binning
  lsh_kernel<<<(BNROWS + 255) / 256, 256, 0, stream>>>(xdist, rot, msk, bin_idx);
  sort_kernel<<<BB, SORT_T, 0, stream>>>(bin_idx, flat);

  // per-bin Gaussian kernel + message passing + scatter
  bin_kernel<<<BB * NBINS, 256, 0, stream>>>(xdist, fhpb, gateb, fhetb, flat,
                                             msk, (float*)d_out);
}
